// GATLSTM_28827820491376
// MI455X (gfx1250) — matrix operand order repaired and checked by reading, not compiled, not run on hardware
//
#include <hip/hip_runtime.h>
#include <hip/hip_bf16.h>

// ---------------------------------------------------------------------------
// GAT (in_ch=1, heads=1) -> LSTM(1600) x2 -> Linear(400), B=16 S=64 N=400 C=4
// bf16 WMMA (v_wmma_f32_16x16x32_bf16) for all GEMMs, f32 accumulation.
// ---------------------------------------------------------------------------

typedef __attribute__((ext_vector_type(16))) __bf16 v16bf;
typedef __attribute__((ext_vector_type(8)))  __bf16 v8bf;
typedef __attribute__((ext_vector_type(8)))  float  v8f;

#define BB   16
#define SS   64
#define NN   400
#define CC   4
#define E0   12800
#define DD   1600      // N*C
#define G4   6400      // 4*D

static __device__ __forceinline__ v16bf load_a_tile(const __hip_bfloat16* rowp,
                                                    int k, int half) {
  // A 16x32 bf16 layout: lane half h holds K in [k+16h, k+16h+16)
  // (two 16B chunks).
  union { v16bf v; v8bf h[2]; } u;
  u.h[0] = *(const v8bf*)(rowp + k + 16 * half);
  u.h[1] = *(const v8bf*)(rowp + k + 16 * half + 8);
  return u.v;
}

static __device__ __forceinline__ v8f wmma_bf16(v16bf a, v16bf b, v8f c) {
  return __builtin_amdgcn_wmma_f32_16x16x32_bf16(false, a, false, b,
                                                 (short)0, c, false, false);
}

static __device__ __forceinline__ float sigmf(float x) {
  return 1.0f / (1.0f + expf(-x));
}

// order-preserving float<->uint for atomic max with negatives
static __device__ __forceinline__ unsigned encf(float v) {
  unsigned u = __float_as_uint(v);
  return (u & 0x80000000u) ? ~u : (u | 0x80000000u);
}
static __device__ __forceinline__ float decf(unsigned u) {
  u = (u & 0x80000000u) ? (u & 0x7FFFFFFFu) : ~u;
  return __uint_as_float(u);
}

// ------------------------------ conversions --------------------------------
__global__ void cvt_f32_bf16(const float* __restrict__ src,
                             __hip_bfloat16* __restrict__ dst, int n) {
  int i = blockIdx.x * blockDim.x + threadIdx.x;
  int stride = gridDim.x * blockDim.x;
  for (; i < n; i += stride) dst[i] = __float2bfloat16(src[i]);
}

__global__ void zero_state(__hip_bfloat16* __restrict__ h,
                           float* __restrict__ c, int n) {
  int i = blockIdx.x * blockDim.x + threadIdx.x;
  if (i < n) { h[i] = __float2bfloat16(0.0f); c[i] = 0.0f; }
}

// --------------------------------- GAT -------------------------------------
// One block per (b,s) graph.  Since in_channels==1:
//   a_src[n] = ws*x[n], a_dst[n] = wd*x[n],  msg agg = (sum ex*x[src])/den * W
__global__ void gat_kernel(const float* __restrict__ x,
                           const int* __restrict__ ei,
                           const float* __restrict__ Wg,
                           const float* __restrict__ att_src,
                           const float* __restrict__ att_dst,
                           const float* __restrict__ bias,
                           __hip_bfloat16* __restrict__ xg) {
  __shared__ float    xs[NN];
  __shared__ unsigned mxs[NN];
  __shared__ float    dens[NN];
  __shared__ float    nums[NN];

  int bs = blockIdx.x;
  int tid = threadIdx.x;
  const float* xp = x + (size_t)bs * NN;

  for (int i = tid; i < NN; i += blockDim.x) {
    xs[i] = xp[i]; mxs[i] = 0u; dens[i] = 0.0f; nums[i] = 0.0f;
  }
  __syncthreads();

  float ws = 0.0f, wd = 0.0f;
  for (int c = 0; c < CC; ++c) { ws += Wg[c] * att_src[c]; wd += Wg[c] * att_dst[c]; }

  const int E = E0 + NN;  // appended self loops
  // pass 1: segment max of leaky_relu(ws*x[src] + wd*x[dst]) over dst
  for (int e = tid; e < E; e += blockDim.x) {
    int src, dst;
    if (e < E0) { src = ei[e]; dst = ei[E0 + e]; } else { src = dst = e - E0; }
    float v = ws * xs[src] + wd * xs[dst];
    v = (v >= 0.0f) ? v : 0.2f * v;
    atomicMax(&mxs[dst], encf(v));
  }
  __syncthreads();
  // pass 2: exp-sum and weighted numerator
  for (int e = tid; e < E; e += blockDim.x) {
    int src, dst;
    if (e < E0) { src = ei[e]; dst = ei[E0 + e]; } else { src = dst = e - E0; }
    float v = ws * xs[src] + wd * xs[dst];
    v = (v >= 0.0f) ? v : 0.2f * v;
    float ex = expf(v - decf(mxs[dst]));
    atomicAdd(&dens[dst], ex);
    atomicAdd(&nums[dst], ex * xs[src]);
  }
  __syncthreads();
  // pass 3: out[n,c] = relu(agg * W[c] + bias[c]) -> bf16 row [bs, n*4+c]
  for (int n = tid; n < NN; n += blockDim.x) {
    float agg = nums[n] / dens[n];
    __hip_bfloat16* o = xg + (size_t)bs * DD + n * CC;
    #pragma unroll
    for (int c = 0; c < CC; ++c) {
      float v = agg * Wg[c] + bias[c];
      o[c] = __float2bfloat16(v > 0.0f ? v : 0.0f);
    }
  }
}

// ----------------------- XW = X @ W^T + (b0 + b1) --------------------------
// X:[1024,K] bf16 row-major, W:[Ncols,K] bf16 row-major (i.e. B = W^T).
// block = 128 (4 waves); wave -> 64 cols (4 tiles); grid = (1024/16, Ncols/256)
__global__ void gemm_xw(const __hip_bfloat16* __restrict__ X,
                        const __hip_bfloat16* __restrict__ W,
                        const float* __restrict__ b0,
                        const float* __restrict__ b1,
                        float* __restrict__ out, int Ncols, int K) {
  int lane = threadIdx.x & 31;
  int wave = threadIdx.x >> 5;
  int m0 = blockIdx.x * 16;
  int n0 = blockIdx.y * 256 + wave * 64;
  int nr = lane & 15, half = lane >> 4;

  const __hip_bfloat16* A = X + (size_t)(m0 + nr) * K;
  const __hip_bfloat16* Bp[4];
  #pragma unroll
  for (int j = 0; j < 4; ++j)
    Bp[j] = W + (size_t)(n0 + j * 16 + nr) * K + half * 16;

  v8f acc[4] = {};
  for (int k = 0; k < K; k += 32) {
    v16bf a = load_a_tile(A, k, half);
    #pragma unroll
    for (int j = 0; j < 4; ++j)
      acc[j] = wmma_bf16(a, *(const v16bf*)(Bp[j] + k), acc[j]);
  }

  #pragma unroll
  for (int j = 0; j < 4; ++j) {
    int col = n0 + j * 16 + nr;
    float bsum = b0[col] + b1[col];
    #pragma unroll
    for (int r = 0; r < 8; ++r) {
      int m = m0 + r + 8 * half;
      out[(size_t)m * Ncols + col] = acc[j][r] + bsum;
    }
  }
}

// ------------------------------ LSTM step ----------------------------------
// g = h_in @ Whh^T + xw[:,t,:];  c,h update.  M=16 (one WMMA tile row).
// block = 128 (4 waves); wave handles one 16-col chunk j across all 4 gates.
// grid = 100 blocks (400 chunks).
__global__ void lstm_step(const __hip_bfloat16* __restrict__ h_in,
                          const __hip_bfloat16* __restrict__ Whh,
                          const float* __restrict__ xw,
                          float* __restrict__ c_state,
                          __hip_bfloat16* __restrict__ h_out,
                          __hip_bfloat16* __restrict__ h_seq,  // may be null
                          int t) {
  int lane = threadIdx.x & 31;
  int wave = threadIdx.x >> 5;
  int j = blockIdx.x * 4 + wave;         // 0..399
  int nr = lane & 15, half = lane >> 4;

  const __hip_bfloat16* A = h_in + (size_t)nr * DD;
  const __hip_bfloat16* Bp[4];
  #pragma unroll
  for (int g = 0; g < 4; ++g)
    Bp[g] = Whh + (size_t)(g * DD + j * 16 + nr) * DD + half * 16;

  v8f acc[4] = {};
  for (int k = 0; k < DD; k += 32) {
    v16bf a = load_a_tile(A, k, half);
    #pragma unroll
    for (int g = 0; g < 4; ++g)
      acc[g] = wmma_bf16(a, *(const v16bf*)(Bp[g] + k), acc[g]);
  }

  int col = j * 16 + nr;
  #pragma unroll
  for (int r = 0; r < 8; ++r) {
    int m = r + 8 * half;                       // batch index 0..15
    size_t row = (size_t)(m * SS + t) * G4;     // xw row for (m, t)
    float gi = acc[0][r] + xw[row + 0 * DD + col];
    float gf = acc[1][r] + xw[row + 1 * DD + col];
    float gg = acc[2][r] + xw[row + 2 * DD + col];
    float go = acc[3][r] + xw[row + 3 * DD + col];
    float cp = c_state[(size_t)m * DD + col];
    float cn = sigmf(gf) * cp + sigmf(gi) * tanhf(gg);
    float hh = sigmf(go) * tanhf(cn);
    c_state[(size_t)m * DD + col] = cn;
    __hip_bfloat16 hb = __float2bfloat16(hh);
    h_out[(size_t)m * DD + col] = hb;
    if (h_seq) h_seq[(size_t)(m * SS + t) * DD + col] = hb;
  }
}

// --------------------- out = h2 @ W_out^T + b_out --------------------------
// M=16, N=400 (25 tiles), K=1600.  grid = 7 blocks x 128 (28 waves, 25 used).
__global__ void out_gemm(const __hip_bfloat16* __restrict__ h2,
                         const __hip_bfloat16* __restrict__ Wout,
                         const float* __restrict__ bout,
                         float* __restrict__ out) {
  int lane = threadIdx.x & 31;
  int wave = threadIdx.x >> 5;
  int tile = blockIdx.x * 4 + wave;
  if (tile >= 25) return;                 // wave-uniform: EXEC stays full
  int nr = lane & 15, half = lane >> 4;

  const __hip_bfloat16* A = h2 + (size_t)nr * DD;
  const __hip_bfloat16* B = Wout + (size_t)(tile * 16 + nr) * DD + half * 16;

  v8f acc = {};
  for (int k = 0; k < DD; k += 32)
    acc = wmma_bf16(load_a_tile(A, k, half), *(const v16bf*)(B + k), acc);

  int col = tile * 16 + nr;
  float bv = bout[col];
  #pragma unroll
  for (int r = 0; r < 8; ++r)
    out[(size_t)(r + 8 * half) * NN + col] = acc[r] + bv;
}

// ------------------------------- launcher ----------------------------------
extern "C" void kernel_launch(void* const* d_in, const int* in_sizes, int n_in,
                              void* d_out, int out_size, void* d_ws, size_t ws_size,
                              hipStream_t stream) {
  const float* x    = (const float*)d_in[0];
  const int*   ei   = (const int*)d_in[1];
  const float* Wg   = (const float*)d_in[2];
  const float* asr  = (const float*)d_in[3];
  const float* ads  = (const float*)d_in[4];
  const float* bg   = (const float*)d_in[5];
  const float* Wih0 = (const float*)d_in[6];
  const float* Whh0 = (const float*)d_in[7];
  const float* bih0 = (const float*)d_in[8];
  const float* bhh0 = (const float*)d_in[9];
  const float* Wih1 = (const float*)d_in[10];
  const float* Whh1 = (const float*)d_in[11];
  const float* bih1 = (const float*)d_in[12];
  const float* bhh1 = (const float*)d_in[13];
  const float* Wout = (const float*)d_in[14];
  const float* bout = (const float*)d_in[15];
  float* out = (float*)d_out;

  // workspace carve-up (256B aligned)
  char* p = (char*)d_ws;
  auto alloc = [&](size_t bytes) -> char* {
    char* r = p; p += (bytes + 255) & ~(size_t)255; return r;
  };
  const size_t WBYTES = (size_t)G4 * DD * sizeof(__hip_bfloat16);
  __hip_bfloat16* wih0b = (__hip_bfloat16*)alloc(WBYTES);
  __hip_bfloat16* whh0b = (__hip_bfloat16*)alloc(WBYTES);
  __hip_bfloat16* wih1b = (__hip_bfloat16*)alloc(WBYTES);
  __hip_bfloat16* whh1b = (__hip_bfloat16*)alloc(WBYTES);
  __hip_bfloat16* woutb = (__hip_bfloat16*)alloc((size_t)NN * DD * 2);
  __hip_bfloat16* xgatb = (__hip_bfloat16*)alloc((size_t)BB * SS * DD * 2);
  __hip_bfloat16* h1seq = (__hip_bfloat16*)alloc((size_t)BB * SS * DD * 2);
  float* xw0 = (float*)alloc((size_t)BB * SS * G4 * 4);
  float* xw1 = (float*)alloc((size_t)BB * SS * G4 * 4);
  __hip_bfloat16* hping = (__hip_bfloat16*)alloc((size_t)BB * DD * 2);
  __hip_bfloat16* hpong = (__hip_bfloat16*)alloc((size_t)BB * DD * 2);
  float* cstate = (float*)alloc((size_t)BB * DD * 4);

  const int NW = G4 * DD;
  cvt_f32_bf16<<<1024, 256, 0, stream>>>(Wih0, wih0b, NW);
  cvt_f32_bf16<<<1024, 256, 0, stream>>>(Whh0, whh0b, NW);
  cvt_f32_bf16<<<1024, 256, 0, stream>>>(Wih1, wih1b, NW);
  cvt_f32_bf16<<<1024, 256, 0, stream>>>(Whh1, whh1b, NW);
  cvt_f32_bf16<<<1024, 256, 0, stream>>>(Wout, woutb, NN * DD);

  gat_kernel<<<BB * SS, 256, 0, stream>>>(x, ei, Wg, asr, ads, bg, xgatb);

  // layer 0
  gemm_xw<<<dim3(BB * SS / 16, G4 / 256), 128, 0, stream>>>(
      xgatb, wih0b, bih0, bhh0, xw0, G4, DD);
  zero_state<<<(BB * DD + 255) / 256, 256, 0, stream>>>(hping, cstate, BB * DD);
  __hip_bfloat16 *hin = hping, *hout = hpong;
  for (int t = 0; t < SS; ++t) {
    lstm_step<<<100, 128, 0, stream>>>(hin, whh0b, xw0, cstate, hout, h1seq, t);
    __hip_bfloat16* tmp = hin; hin = hout; hout = tmp;
  }

  // layer 1
  gemm_xw<<<dim3(BB * SS / 16, G4 / 256), 128, 0, stream>>>(
      h1seq, wih1b, bih1, bhh1, xw1, G4, DD);
  zero_state<<<(BB * DD + 255) / 256, 256, 0, stream>>>(hping, cstate, BB * DD);
  hin = hping; hout = hpong;
  for (int t = 0; t < SS; ++t) {
    lstm_step<<<100, 128, 0, stream>>>(hin, whh1b, xw1, cstate, hout,
                                       (__hip_bfloat16*)nullptr, t);
    __hip_bfloat16* tmp = hin; hin = hout; hout = tmp;
  }

  // final projection (hin holds h2[:, S-1, :])
  out_gemm<<<7, 128, 0, stream>>>(hin, woutb, bout, out);
}
